// MultiHeadAttention_39402029973761
// MI455X (gfx1250) — matrix operand order repaired and checked by reading, not compiled, not run on hardware
//
#include <hip/hip_runtime.h>

// ---------------- Types ----------------
typedef __attribute__((ext_vector_type(16))) __bf16 v16bf;
typedef __attribute__((ext_vector_type(8)))  float  v8f;
typedef __attribute__((ext_vector_type(4)))  unsigned int u32x4;

#define S_LEN   2048
#define DMODEL  1024
#define NHEADS  16
#define DK      64
#define BATCH   2

// ---------------- WMMA wrapper ----------------
__device__ __forceinline__ v8f wmma_bf16(v16bf a, v16bf b, v8f c) {
  return __builtin_amdgcn_wmma_f32_16x16x32_bf16(
      false, a, false, b, (short)0, c, false, false);
}

__device__ __forceinline__ v8f zero8() {
  v8f z;
#pragma unroll
  for (int i = 0; i < 8; ++i) z[i] = 0.0f;
  return z;
}

// ---------------- LDS helpers / async copy ----------------
__device__ __forceinline__ unsigned lds_off(const void* p) {
  // generic -> LDS(AS3) addrspace cast; AS3 pointers are 32-bit LDS addresses
  return (unsigned)(size_t)(const __attribute__((address_space(3))) void*)p;
}

// One 16B global->LDS async copy per lane (tracked by ASYNCcnt)
__device__ __forceinline__ void async_copy16(const void* gsrc, unsigned ldst) {
  asm volatile("global_load_async_to_lds_b128 %0, %1, off"
               :: "v"(ldst), "v"(gsrc) : "memory");
}

#define WAIT_ASYNC(n) asm volatile("s_wait_asynccnt " #n ::: "memory")

// ---------------- Fragment loaders ----------------
// A-matrix 16x32 bf16: lane l -> row m=l&15, half h=l>>4.
// elements 0..7 = A[m][8h..8h+7], elements 8..15 = A[m][16+8h..16+8h+7]
__device__ __forceinline__ v16bf load_a_frag(const __bf16* src, int ld) {
  const int lane = threadIdx.x & 31;
  const int m = lane & 15, h = (lane >> 4) & 1;
  const __bf16* p0 = src + (size_t)m * ld + 8 * h;
  union { u32x4 q[2]; v16bf v; } u;
  u.q[0] = *(const u32x4*)p0;
  u.q[1] = *(const u32x4*)(p0 + 16);
  return u.v;
}

// Same fragment but f32 source (convert to bf16 in-register)
__device__ __forceinline__ v16bf load_a_frag(const float* src, int ld) {
  const int lane = threadIdx.x & 31;
  const int m = lane & 15, h = (lane >> 4) & 1;
  const float* p0 = src + (size_t)m * ld + 8 * h;
  const float* p1 = p0 + 16;
  v16bf out;
#pragma unroll
  for (int i = 0; i < 8; ++i) out[i] = (__bf16)p0[i];
#pragma unroll
  for (int i = 0; i < 8; ++i) out[i + 8] = (__bf16)p1[i];
  return out;
}

// B-matrix 32x16 bf16: lane l -> col n=l&15, half h=l>>4; elements 0..7 = B[8h..8h+7][n], 8..15 = B[16+8h..16+8h+7][n].
__device__ __forceinline__ v16bf load_b_frag(const __bf16* src, int ld) {
  const int lane = threadIdx.x & 31;
  const int n = lane & 15, h = (lane >> 4) & 1;
  const __bf16* p = src + (size_t)(8 * h) * ld + n;
  v16bf out;
#pragma unroll
  for (int i = 0; i < 8; ++i) out[i] = p[(size_t)i * ld];
#pragma unroll
  for (int i = 0; i < 8; ++i) out[i + 8] = p[(size_t)(16 + i) * ld];
  return out;
}

// ---------------- Weight repack / convert ----------------
__global__ void repack_w_kernel(const float* __restrict__ W, __bf16* __restrict__ out) {
  int idx = blockIdx.x * 256 + threadIdx.x;           // H*D*64 = 1M
  if (idx >= NHEADS * DMODEL * DK) return;
  int k = idx & 63;
  int d = (idx >> 6) & (DMODEL - 1);
  int h = idx >> 16;
  out[(size_t)d * (NHEADS * DK) + h * DK + k] = (__bf16)W[idx];
}

__global__ void cvt_f32_bf16_kernel(const float* __restrict__ in, __bf16* __restrict__ out, int n) {
  int idx = blockIdx.x * 256 + threadIdx.x;
  if (idx < n) out[idx] = (__bf16)in[idx];
}

// ---------------- Generic WMMA GEMM ----------------
// C[M,N] = A[M,K] * B[K,N] + bias[N]; block = 4 waves, tile 64(M) x 64(N).
// B is staged global->LDS in 64-deep K slabs with async copies, double-buffered.
enum { OUT_BHSK = 0, OUT_BHKS = 1, OUT_F32 = 2 };

template <typename AT, int OUT_MODE>
__global__ __launch_bounds__(128) void gemm_wmma_kernel(
    const AT* __restrict__ A, const __bf16* __restrict__ Bm,
    const float* __restrict__ bias, void* __restrict__ Cout,
    int M, int N, int K)
{
  const int w = threadIdx.x >> 5;
  const int lane = threadIdx.x & 31;
  const int row_base = blockIdx.x * 64 + w * 16;
  const int n_base = blockIdx.y * 64;

  __shared__ __bf16 btile[2][64][64];   // 16 KB double-buffered B slab

  // stage slab s (k rows s*64..s*64+63) into buffer bi: 4 async ops per wave
  auto stage_b = [&](int s, int bi) {
#pragma unroll
    for (int i = 0; i < 4; ++i) {
      int row = w * 16 + i * 4 + (lane >> 3);
      const __bf16* g = Bm + (size_t)(s * 64 + row) * N + n_base + (lane & 7) * 8;
      async_copy16((const void*)g, lds_off(&btile[bi][row][(lane & 7) * 8]));
    }
  };

  v8f acc[4];
#pragma unroll
  for (int nf = 0; nf < 4; ++nf) acc[nf] = zero8();

  const int nslab = K / 64;
  stage_b(0, 0);
  for (int s = 0; s < nslab; ++s) {
    const int bi = s & 1;
    if (s + 1 < nslab) { stage_b(s + 1, bi ^ 1); WAIT_ASYNC(0x4); }
    else               { WAIT_ASYNC(0x0); }
    __syncthreads();
#pragma unroll
    for (int half = 0; half < 2; ++half) {
      v16bf a = load_a_frag(A + (size_t)row_base * K + s * 64 + half * 32, K);
#pragma unroll
      for (int nf = 0; nf < 4; ++nf) {
        v16bf b = load_b_frag(&btile[bi][half * 32][nf * 16], 64);
        acc[nf] = wmma_bf16(a, b, acc[nf]);
      }
    }
    __syncthreads();   // all waves done reading before buffer is overwritten
  }

  const int n_lo = lane & 15;
  const int rgrp = (lane >> 4) * 8;
#pragma unroll
  for (int nf = 0; nf < 4; ++nf) {
#pragma unroll
    for (int v = 0; v < 8; ++v) {
      int r = row_base + rgrp + v;
      int c = n_base + nf * 16 + n_lo;
      float val = acc[nf][v] + bias[c];
      if (OUT_MODE == OUT_F32) {
        ((float*)Cout)[(size_t)r * N + c] = val;
      } else {
        int bb = r >> 11, s2 = r & (S_LEN - 1);    // r = b*S + s
        int hh = c >> 6,  kk = c & 63;             // c = h*64 + k
        size_t idx;
        if (OUT_MODE == OUT_BHSK)
          idx = (((size_t)bb * NHEADS + hh) * S_LEN + s2) * DK + kk;
        else
          idx = (((size_t)bb * NHEADS + hh) * DK + kk) * S_LEN + s2;
        ((__bf16*)Cout)[idx] = (__bf16)val;
      }
    }
  }
}

// ---------------- Flash attention ----------------
// grid: (S/64, B*H); block = 128 (4 waves). Wave w owns 16 q-rows.
// K/V tiles staged global->LDS with async copies, double-buffered.
// Reference: attn = softmax(raw scores) * 1/(2*DK) -> folded into epilogue.
__global__ __launch_bounds__(128) void attn_kernel(
    const __bf16* __restrict__ QW,    // [B*H, S, 64]
    const __bf16* __restrict__ KWT,   // [B*H, 64, S]
    const __bf16* __restrict__ VW,    // [B*H, S, 64]
    __bf16* __restrict__ Concat)      // [B, S, H*64]
{
  const int q0 = blockIdx.x * 64;
  const int bh = blockIdx.y;
  const int b  = bh >> 4;
  const int h  = bh & (NHEADS - 1);
  const int w    = threadIdx.x >> 5;
  const int lane = threadIdx.x & 31;

  __shared__ __bf16 ktile[2][64][64];   // 16 KB: rows = d, cols = t_local
  __shared__ __bf16 vtile[2][64][64];   // 16 KB: rows = t_local, cols = d
  __shared__ __bf16 lds_p[4][16][64];   //  8 KB: per-wave P tile

  const __bf16* kbase = KWT + (size_t)bh * DK * S_LEN;
  const __bf16* vbase = VW + (size_t)bh * S_LEN * DK;

  // stage K and V tiles for t-tile `it` into buffer bi: 8 async ops per wave
  auto stage_kv = [&](int it, int bi) {
    const int t0 = it * 64;
#pragma unroll
    for (int i = 0; i < 4; ++i) {
      int row = w * 16 + i * 4 + (lane >> 3);
      int col8 = (lane & 7) * 8;
      const __bf16* gk = kbase + (size_t)row * S_LEN + t0 + col8;
      async_copy16((const void*)gk, lds_off(&ktile[bi][row][col8]));
      const __bf16* gv = vbase + (size_t)(t0 + row) * DK + col8;
      async_copy16((const void*)gv, lds_off(&vtile[bi][row][col8]));
    }
  };

  // Q fragments stay in registers for the whole kernel
  const __bf16* qbase = QW + (((size_t)bh * S_LEN) + q0 + w * 16) * DK;
  v16bf aq0 = load_a_frag(qbase + 0, DK);
  v16bf aq1 = load_a_frag(qbase + 32, DK);

  v8f o[4];
#pragma unroll
  for (int df = 0; df < 4; ++df) o[df] = zero8();
  float mrow[8], lrow[8];
#pragma unroll
  for (int v = 0; v < 8; ++v) { mrow[v] = -INFINITY; lrow[v] = 0.0f; }

  const int ntile = S_LEN / 64;
  stage_kv(0, 0);
  for (int it = 0; it < ntile; ++it) {
    const int bi = it & 1;
    if (it + 1 < ntile) { stage_kv(it + 1, bi ^ 1); WAIT_ASYNC(0x8); }
    else                { WAIT_ASYNC(0x0); }
    __syncthreads();

    // ---- scores tile S[16 x 64] = Q(16x64) . K^T (B frags from LDS) ----
    v8f s[4];
#pragma unroll
    for (int nf = 0; nf < 4; ++nf) s[nf] = zero8();
#pragma unroll
    for (int kk = 0; kk < 2; ++kk) {
      v16bf aq = (kk == 0) ? aq0 : aq1;
#pragma unroll
      for (int nf = 0; nf < 4; ++nf)
        s[nf] = wmma_bf16(aq, load_b_frag(&ktile[bi][kk * 32][nf * 16], 64), s[nf]);
    }

    // ---- online softmax update (row stats per (half, v)) ----
    float mnew[8], alpha[8];
#pragma unroll
    for (int v = 0; v < 8; ++v) {
      float mx = fmaxf(fmaxf(s[0][v], s[1][v]), fmaxf(s[2][v], s[3][v]));
      mx = fmaxf(mx, __shfl_xor(mx, 1));
      mx = fmaxf(mx, __shfl_xor(mx, 2));
      mx = fmaxf(mx, __shfl_xor(mx, 4));
      mx = fmaxf(mx, __shfl_xor(mx, 8));
      mnew[v] = fmaxf(mrow[v], mx);
      alpha[v] = __expf(mrow[v] - mnew[v]);
      mrow[v] = mnew[v];
    }
    const int rg = (lane >> 4) * 8;
    const int nl = lane & 15;
#pragma unroll
    for (int v = 0; v < 8; ++v) {
      float sum = 0.0f;
#pragma unroll
      for (int nf = 0; nf < 4; ++nf) {
        float p = __expf(s[nf][v] - mnew[v]);
        sum += p;
        lds_p[w][rg + v][nf * 16 + nl] = (__bf16)p;
      }
      sum += __shfl_xor(sum, 1);
      sum += __shfl_xor(sum, 2);
      sum += __shfl_xor(sum, 4);
      sum += __shfl_xor(sum, 8);
      lrow[v] = lrow[v] * alpha[v] + sum;
#pragma unroll
      for (int df = 0; df < 4; ++df) o[df][v] *= alpha[v];
    }
    // per-wave LDS region; make P stores visible before re-reading as A frags
    asm volatile("s_wait_dscnt 0" ::: "memory");

    // ---- O += P(16x64) . V(64x64) (both operands from LDS) ----
#pragma unroll
    for (int kk = 0; kk < 2; ++kk) {
      v16bf ap = load_a_frag(&lds_p[w][0][kk * 32], 64);
#pragma unroll
      for (int df = 0; df < 4; ++df)
        o[df] = wmma_bf16(ap, load_b_frag(&vtile[bi][kk * 32][df * 16], 64), o[df]);
    }
    __syncthreads();   // all waves done with ktile/vtile[bi] before overwrite
  }

  // ---- epilogue: normalize by rowsum, apply reference's 1/(2*DK) ----
  const int rg = (lane >> 4) * 8;
  const int nl = lane & 15;
#pragma unroll
  for (int df = 0; df < 4; ++df) {
#pragma unroll
    for (int v = 0; v < 8; ++v) {
      float val = o[df][v] / (lrow[v] * (2.0f * (float)DK));
      int srow = q0 + w * 16 + rg + v;
      size_t idx = (((size_t)b * S_LEN) + srow) * (NHEADS * DK) + h * DK + df * 16 + nl;
      Concat[idx] = (__bf16)val;
    }
  }
}

// ---------------- Host launcher ----------------
extern "C" void kernel_launch(void* const* d_in, const int* in_sizes, int n_in,
                              void* d_out, int out_size, void* d_ws, size_t ws_size,
                              hipStream_t stream) {
  const float* Q  = (const float*)d_in[0];
  const float* K  = (const float*)d_in[1];
  const float* V  = (const float*)d_in[2];
  const float* Wq = (const float*)d_in[3];
  const float* bq = (const float*)d_in[4];
  const float* Wk = (const float*)d_in[5];
  const float* bk = (const float*)d_in[6];
  const float* Wv = (const float*)d_in[7];
  const float* bv = (const float*)d_in[8];
  const float* Wo = (const float*)d_in[9];
  const float* bo = (const float*)d_in[10];

  char* ws = (char*)d_ws;
  const size_t MB = 1024ull * 1024ull;
  __bf16* Wqb    = (__bf16*)(ws + 0 * MB);   // 2 MB each
  __bf16* Wkb    = (__bf16*)(ws + 2 * MB);
  __bf16* Wvb    = (__bf16*)(ws + 4 * MB);
  __bf16* Wob    = (__bf16*)(ws + 6 * MB);
  __bf16* QWb    = (__bf16*)(ws + 8 * MB);   // 8 MB each
  __bf16* KWT    = (__bf16*)(ws + 16 * MB);
  __bf16* VWb    = (__bf16*)(ws + 24 * MB);
  __bf16* Concat = (__bf16*)(ws + 32 * MB);  // ends at 40 MB

  const int M = BATCH * S_LEN;        // 4096
  const int N = NHEADS * DK;          // 1024
  const int Kd = DMODEL;              // 1024
  const int wn = NHEADS * DMODEL * DK;

  // 1) weight repack / convert
  repack_w_kernel<<<(wn + 255) / 256, 256, 0, stream>>>(Wq, Wqb);
  repack_w_kernel<<<(wn + 255) / 256, 256, 0, stream>>>(Wk, Wkb);
  repack_w_kernel<<<(wn + 255) / 256, 256, 0, stream>>>(Wv, Wvb);
  cvt_f32_bf16_kernel<<<(DMODEL * DMODEL + 255) / 256, 256, 0, stream>>>(Wo, Wob, DMODEL * DMODEL);

  // 2) projections (A read as f32, converted in-register)
  dim3 gproj(M / 64, N / 64);
  gemm_wmma_kernel<float, OUT_BHSK><<<gproj, 128, 0, stream>>>(Q, Wqb, bq, (void*)QWb, M, N, Kd);
  gemm_wmma_kernel<float, OUT_BHKS><<<gproj, 128, 0, stream>>>(K, Wkb, bk, (void*)KWT, M, N, Kd);
  gemm_wmma_kernel<float, OUT_BHSK><<<gproj, 128, 0, stream>>>(V, Wvb, bv, (void*)VWb, M, N, Kd);

  // 3) flash attention
  dim3 gattn(S_LEN / 64, BATCH * NHEADS);
  attn_kernel<<<gattn, 128, 0, stream>>>(QWb, KWT, VWb, Concat);

  // 4) output projection -> f32 d_out
  gemm_wmma_kernel<__bf16, OUT_F32><<<gproj, 128, 0, stream>>>(Concat, Wob, bo, d_out, M, DMODEL, N);
}
